// Latent_SSM_Structured_Mesh_2D_23845658428253
// MI455X (gfx1250) — compile-verified, matrix operand order checked by reading, not run on hardware
//
#include <hip/hip_runtime.h>
#include <hip/hip_bf16.h>
#include <cstdint>

typedef __attribute__((ext_vector_type(16))) _Float16 v16h;
typedef __attribute__((ext_vector_type(8)))  float    v8f;

#define B_      16
#define HEADS   8
#define G_      64
#define DH      64
#define NSEQ    3131
#define DM      512
#define NH      8
#define CONVDIM 514
#define DINPROJ 1042
#define DIM_    512

__device__ __forceinline__ float sigmoidf_(float x) { return 1.0f / (1.0f + __expf(-x)); }
__device__ __forceinline__ float softplusf_(float x) { return (x > 20.0f) ? x : log1pf(__expf(x)); }

// -------- Kernel 1: zxbcdt = u @ w_in.T, u gathered from slice_token ----------
__global__ __launch_bounds__(256) void k_proj(const float* __restrict__ tok,
                                              const float* __restrict__ w_in,
                                              float* __restrict__ zx) {
    __shared__ float u[DM];
    int b = blockIdx.x >> 6;
    int g = blockIdx.x & 63;
    int tid = threadIdx.x;
    for (int i = tid; i < DM; i += 256) {
        int h = i >> 6, c = i & 63;
        u[i] = tok[(((size_t)(b * HEADS + h)) * G_ + g) * DH + c];
    }
    __syncthreads();
    for (int o = tid; o < DINPROJ; o += 256) {
        const float4* w4 = (const float4*)(w_in + (size_t)o * DM);
        float acc = 0.f;
#pragma unroll 4
        for (int k = 0; k < DM / 4; ++k) {
            float4 w = w4[k];
            acc += u[4 * k] * w.x + u[4 * k + 1] * w.y + u[4 * k + 2] * w.z + u[4 * k + 3] * w.w;
        }
        zx[((size_t)(b * G_ + g)) * DINPROJ + o] = acc;
    }
}

// -------- Kernel 2: causal depthwise conv (DCONV=3) + SiLU on xBC channels ----
__global__ __launch_bounds__(256) void k_conv(const float* __restrict__ zx,
                                              const float* __restrict__ conv_w,
                                              const float* __restrict__ conv_b,
                                              float* __restrict__ xbc) {
    int i = blockIdx.x * 256 + threadIdx.x;
    if (i >= B_ * G_ * CONVDIM) return;
    int c = i % CONVDIM;
    int t = (i / CONVDIM) % G_;
    int b = i / (CONVDIM * G_);
    float acc = conv_b[c];
#pragma unroll
    for (int k = 0; k < 3; ++k) {
        int tt = t + k - 2;
        if (tt >= 0) acc += conv_w[c * 3 + k] * zx[((size_t)(b * G_ + tt)) * DINPROJ + DM + c];
    }
    xbc[(size_t)i] = acc * sigmoidf_(acc);
}

// -------- Kernel 3: bidirectional SSM scan (fw + time-reversed bw), roll-by-1 -
__global__ __launch_bounds__(64) void k_scan(const float* __restrict__ zx,
                                             const float* __restrict__ xbc,
                                             const float* __restrict__ dt_bias,
                                             const float* __restrict__ A_log,
                                             float* __restrict__ ycore) {
    __shared__ float Bv[G_], Cv[G_], dtf[G_], dtb[G_];
    __shared__ float ybuf[G_][DH];  // [t][c], each thread owns column c
    int h = blockIdx.x & 7;
    int b = blockIdx.x >> 3;
    int c = threadIdx.x;
    {
        int t = c;
        const float* row = xbc + ((size_t)(b * G_ + t)) * CONVDIM;
        Bv[t] = row[DM];
        Cv[t] = row[DM + 1];
        const float* zr = zx + ((size_t)(b * G_ + t)) * DINPROJ;
        float bias = dt_bias[h];
        dtf[t] = softplusf_(zr[DM + CONVDIM + h] + bias);
        dtb[t] = softplusf_(zr[DM + CONVDIM + NH + h] + bias);
    }
    __syncthreads();
    float An = -__expf(A_log[h]);
    // forward: contribution of state(t-1)*C(t-1) lands at output time t
    float st = 0.f, prev = 0.f;
    for (int t = 0; t < G_; ++t) {
        ybuf[t][c] = prev;
        float xv = xbc[((size_t)(b * G_ + t)) * CONVDIM + h * DH + c];
        st = __expf(dtf[t] * An) * st + dtf[t] * Bv[t] * xv;
        prev = st * Cv[t];
    }
    // backward: state over source times 63..t, times C(t), lands at output t-1
    st = 0.f;
    for (int t = G_ - 1; t >= 1; --t) {
        float xv = xbc[((size_t)(b * G_ + t)) * CONVDIM + h * DH + c];
        st = __expf(dtb[t] * An) * st + dtb[t] * Bv[t] * xv;
        ybuf[t - 1][c] += st * Cv[t];
    }
    for (int t = 0; t < G_; ++t)
        ycore[((size_t)(b * G_ + t)) * DM + h * DH + c] = ybuf[t][c];
}

// -------- Kernel 4: +D-term, *silu(z), RMSNorm, @w_out.T; writes outp layout --
__global__ __launch_bounds__(256) void k_norm(const float* __restrict__ zx,
                                              const float* __restrict__ xbc,
                                              const float* __restrict__ ycore,
                                              const float* __restrict__ fc_D_w,
                                              const float* __restrict__ Dp,
                                              const float* __restrict__ norm_w,
                                              const float* __restrict__ w_out,
                                              float* __restrict__ outp) {
    __shared__ float xog[DM], yv[DM], red[256], dco[NH];
    __shared__ float rmss;
    int b = blockIdx.x >> 6, t = blockIdx.x & 63;
    int tid = threadIdx.x;
    const float* xrow = xbc + ((size_t)(b * G_ + t)) * CONVDIM;
    for (int i = tid; i < DM; i += 256) xog[i] = xrow[i];
    __syncthreads();
    {   // dcoef[h] = x_og . fc_D_w[h] + D[h]   (8 groups of 32 lanes)
        int h = tid >> 5, lane = tid & 31;
        float s = 0.f;
        for (int k = lane; k < DM; k += 32) s += xog[k] * fc_D_w[(size_t)h * DM + k];
        red[tid] = s;
    }
    __syncthreads();
    if (tid < NH) {
        float s = 0.f;
        for (int j = 0; j < 32; ++j) s += red[tid * 32 + j];
        dco[tid] = s + Dp[tid];
    }
    __syncthreads();
    const float* zr = zx + ((size_t)(b * G_ + t)) * DINPROJ;
    const float* yr = ycore + ((size_t)(b * G_ + t)) * DM;
    float ss = 0.f;
    for (int d = tid; d < DM; d += 256) {
        float y = yr[d] + xog[d] * dco[d >> 6];
        float z = zr[d];
        y *= z * sigmoidf_(z);
        yv[d] = y;
        ss += y * y;
    }
    red[tid] = ss;
    __syncthreads();
    for (int s = 128; s > 0; s >>= 1) {
        if (tid < s) red[tid] += red[tid + s];
        __syncthreads();
    }
    if (tid == 0) rmss = rsqrtf(red[0] / (float)DM + 1e-5f);
    __syncthreads();
    for (int d = tid; d < DM; d += 256) yv[d] = yv[d] * rmss * norm_w[d];
    __syncthreads();
    for (int d = tid; d < DM; d += 256) {
        const float4* w4 = (const float4*)(w_out + (size_t)d * DM);
        float acc = 0.f;
#pragma unroll 4
        for (int k = 0; k < DM / 4; ++k) {
            float4 w = w4[k];
            acc += yv[4 * k] * w.x + yv[4 * k + 1] * w.y + yv[4 * k + 2] * w.z + yv[4 * k + 3] * w.w;
        }
        int h = d >> 6, c = d & 63;
        outp[(((size_t)(b * HEADS + h)) * G_ + t) * DH + c] = acc;  // (b,h,g,c) layout
    }
}

// -------- Kernel 5: M2[b, h*64+g, d] = sum_c outp[b,h,g,c]*to_out_w[d,h*64+c] (f16)
__global__ __launch_bounds__(256) void k_m2(const float* __restrict__ outp,
                                            const float* __restrict__ to_out_w,
                                            _Float16* __restrict__ m2) {
    __shared__ float op[G_][DH];
    __shared__ float wt[128][DH + 1];
    int b = blockIdx.z, h = blockIdx.y, d0 = blockIdx.x * 128;
    int tid = threadIdx.x;
    for (int i = tid; i < G_ * DH; i += 256) {
        int g = i >> 6, c = i & 63;
        op[g][c] = outp[(((size_t)(b * HEADS + h)) * G_ + g) * DH + c];
    }
    for (int i = tid; i < 128 * DH; i += 256) {
        int d = i >> 6, c = i & 63;
        wt[d][c] = to_out_w[(size_t)(d0 + d) * DM + h * DH + c];
    }
    __syncthreads();
    for (int i = tid; i < G_ * 128; i += 256) {
        int d = i & 127, g = i >> 7;
        float acc = 0.f;
#pragma unroll 8
        for (int c = 0; c < DH; ++c) acc += op[g][c] * wt[d][c];
        m2[((size_t)b * DM + (h * G_ + g)) * DIM_ + d0 + d] = (_Float16)acc;
    }
}

// -------- Kernel 6: batched WMMA GEMM: out[b,n,d] = W(n,hg) @ M2(hg,d) + bias --
#define BM 128
#define BN 128
#define BK 32
#define LP 36  // LDS row pitch in halves (bank-spread padding)

// ISA 7.12.2 16-bit A fragment: lane row = l&15, k-half = l>>4; VGPR v holds
// K = (v/4)*16 + (l>>4)*8 + (v%4)*2 (packed pair). Same loader used for B via
// the K-transposed LDS tile.
__device__ __forceinline__ v16h load_frag(const _Float16* row, int kh) {
    v16h f;
#pragma unroll
    for (int v = 0; v < 8; ++v) {
        int kb = ((v >> 2) << 4) + (kh << 3) + ((v & 3) << 1);
        f[2 * v] = row[kb];
        f[2 * v + 1] = row[kb + 1];
    }
    return f;
}

__global__ __launch_bounds__(256) void k_gemm(const float* __restrict__ sw,
                                              const _Float16* __restrict__ m2,
                                              const float* __restrict__ bias,
                                              float* __restrict__ out) {
    __shared__ _Float16 As[BM * LP];  // [n_row][k]
    __shared__ _Float16 Bs[BN * LP];  // [d_col][k]  (K-transposed)
    int b = blockIdx.z;
    int n0 = blockIdx.y * BM;
    int d0 = blockIdx.x * BN;
    int tid = threadIdx.x;
    int wave = tid >> 5, lane = tid & 31;
    int wr = wave & 3, wc = wave >> 2;  // wave patch: rows 32*wr, cols 64*wc
    int r = lane & 15, kh = lane >> 4;

    v8f acc[2][4] = {};

    for (int k0 = 0; k0 < DM; k0 += BK) {
        // A tile: slice_weights[b,h,n,g] with k = h*64+g; BK=32 keeps h fixed.
        int h = k0 >> 6, g0 = k0 & 63;
        const float* abase = sw + (((size_t)(b * HEADS + h)) * NSEQ + n0) * 64 + g0;
        for (int i = tid; i < BM * BK / 4; i += 256) {
            int row = i >> 3, c4 = (i & 7) << 2;
            float4 v = {0.f, 0.f, 0.f, 0.f};
            if (n0 + row < NSEQ) v = *(const float4*)(abase + (size_t)row * 64 + c4);
            _Float16* dst = &As[row * LP + c4];
            dst[0] = (_Float16)v.x; dst[1] = (_Float16)v.y;
            dst[2] = (_Float16)v.z; dst[3] = (_Float16)v.w;
        }
        // B tile: m2[b, k, d] -> Bs[d][k] (transpose into LDS)
        for (int i = tid; i < BK * BN / 8; i += 256) {
            int k = i >> 4, dc = (i & 15) << 3;
            const _Float16* src = m2 + ((size_t)b * DM + (k0 + k)) * DIM_ + d0 + dc;
#pragma unroll
            for (int j = 0; j < 8; ++j) Bs[(dc + j) * LP + k] = src[j];
        }
        __syncthreads();

        v16h af[2], bf[4];
#pragma unroll
        for (int i = 0; i < 2; ++i) af[i] = load_frag(&As[(wr * 32 + i * 16 + r) * LP], kh);
#pragma unroll
        for (int j = 0; j < 4; ++j) bf[j] = load_frag(&Bs[(wc * 64 + j * 16 + r) * LP], kh);
#pragma unroll
        for (int i = 0; i < 2; ++i)
#pragma unroll
            for (int j = 0; j < 4; ++j)
                acc[i][j] = __builtin_amdgcn_wmma_f32_16x16x32_f16(
                    false, af[i], false, bf[j], (short)0, acc[i][j], false, false);
        __syncthreads();
    }

    // D layout: VGPR q -> M = q + (lane>>4)*8 ; N = lane&15
#pragma unroll
    for (int i = 0; i < 2; ++i) {
        int mb = n0 + wr * 32 + i * 16 + kh * 8;
#pragma unroll
        for (int j = 0; j < 4; ++j) {
            int nb = d0 + wc * 64 + j * 16 + r;
            float bv = bias[nb];
#pragma unroll
            for (int q = 0; q < 8; ++q) {
                int m = mb + q;
                if (m < NSEQ) out[((size_t)b * NSEQ + m) * DIM_ + nb] = acc[i][j][q] + bv;
            }
        }
    }
}

extern "C" void kernel_launch(void* const* d_in, const int* in_sizes, int n_in,
                              void* d_out, int out_size, void* d_ws, size_t ws_size,
                              hipStream_t stream) {
    (void)in_sizes; (void)n_in; (void)out_size; (void)ws_size;
    const float* tok      = (const float*)d_in[0];
    const float* sw       = (const float*)d_in[1];
    const float* w_in     = (const float*)d_in[2];
    const float* conv_w   = (const float*)d_in[3];
    const float* conv_b   = (const float*)d_in[4];
    const float* dt_bias  = (const float*)d_in[5];
    const float* A_log    = (const float*)d_in[6];
    const float* Dp       = (const float*)d_in[7];
    const float* fc_D_w   = (const float*)d_in[8];
    const float* norm_w   = (const float*)d_in[9];
    const float* w_out    = (const float*)d_in[10];
    const float* to_out_w = (const float*)d_in[11];
    const float* to_out_b = (const float*)d_in[12];
    float* out = (float*)d_out;

    const size_t FINAL_N = (size_t)B_ * NSEQ * DIM_;    // 25,649,152
    const size_t INP_N   = (size_t)B_ * HEADS * G_ * DH; // 524,288
    float* out_final = out;
    float* out_inp   = out + FINAL_N;
    float* out_outp  = out + FINAL_N + INP_N;

    char* ws = (char*)d_ws;
    float*    zx    = (float*)ws;                                    // 1,067,008 f
    float*    xbc   = (float*)(ws + 4268032);                        //   526,336 f
    float*    ycore = (float*)(ws + 4268032 + 2105344);              //   524,288 f
    _Float16* m2    = (_Float16*)(ws + 4268032 + 2105344 + 2097152); // 4,194,304 h

    k_proj<<<dim3(B_ * G_), dim3(256), 0, stream>>>(tok, w_in, zx);
    k_conv<<<dim3((B_ * G_ * CONVDIM + 255) / 256), dim3(256), 0, stream>>>(zx, conv_w, conv_b, xbc);
    k_scan<<<dim3(B_ * NH), dim3(64), 0, stream>>>(zx, xbc, dt_bias, A_log, ycore);
    k_norm<<<dim3(B_ * G_), dim3(256), 0, stream>>>(zx, xbc, ycore, fc_D_w, Dp, norm_w, w_out, out_outp);
    hipMemcpyAsync(out_inp, tok, INP_N * sizeof(float), hipMemcpyDeviceToDevice, stream);
    k_m2<<<dim3(4, HEADS, B_), dim3(256), 0, stream>>>(out_outp, to_out_w, m2);
    k_gemm<<<dim3(DIM_ / BN, (NSEQ + BM - 1) / BM, B_), dim3(256), 0, stream>>>(sw, m2, to_out_b, out_final);
}
